// MoELayer_7258494730507
// MI455X (gfx1250) — matrix operand order repaired and checked by reading, not compiled, not run on hardware
//
#include <hip/hip_runtime.h>

// ---------------------------------------------------------------------------
// MoE layer for MI455X (gfx1250, wave32, WMMA).
//
// Reference quirk: top-2 expert indices come from token 0 only, so the core is
// two dense GEMMs [16384,2048]@[2048,2048] fused over the 2 selected experts,
// scaled per-token by that token's own top-2 softmax scores (+ biases).
// 275 GFLOP vs ~300MB compulsory traffic => strongly compute-bound; both
// selected expert matrices (32MB) are L2-resident (192MB L2).
//
// Precision: fp32 -> bf16 hi+lo split; hi*hi + hi*lo + lo*hi accumulated in
// fp32 via v_wmma_f32_16x16x32_bf16 (~fp32 accuracy, 3/8 the cost of the
// 16x16x4 f32 WMMA pipe).
//
// Pipeline (ws_size permitting):
//   1) gate kernel: softmax gates, per-token top-2 scores, token-0 indices
//   2) w_convert kernel: selected W -> split bf16, pre-tiled to the exact LDS
//      fragment layout (one-time pass; avoids 128x redundant converts)
//   3) GEMM kernel: B tiles staged with global_load_async_to_lds_b128
//      (ASYNCcnt), A tiles converted in-register; 96 WMMAs per K-chunk of 64.
// ---------------------------------------------------------------------------

#define N_TOK   16384
#define D_IN    2048
#define D_HID   2048
#define N_EXP   8

typedef __attribute__((ext_vector_type(16))) __bf16 v16bf;
typedef __attribute__((ext_vector_type(8)))  float  v8f;

// ---- workspace layout ------------------------------------------------------
#define WS_SCORE_OFF   256                                  // [N_TOK][2] f32
#define WS_WHI_OFF     ((size_t)1 << 20)
#define WS_WSPLIT_SZ   ((size_t)2 * D_IN * D_HID * 2)       // 16.78 MB each
#define WS_WLO_OFF     (WS_WHI_OFF + WS_WSPLIT_SZ)
#define WS_NEED        (WS_WLO_OFF + WS_WSPLIT_SZ)

// ---- fp32 -> bf16 (RNE) split helpers --------------------------------------
__device__ __forceinline__ unsigned short f32_to_bf16(float f) {
    unsigned u = __float_as_uint(f);
    unsigned r = u + 0x7FFFu + ((u >> 16) & 1u);
    return (unsigned short)(r >> 16);
}
__device__ __forceinline__ void cvt_split(float f, unsigned short& hi, unsigned short& lo) {
    hi = f32_to_bf16(f);
    float hf = __uint_as_float(((unsigned)hi) << 16);
    lo = f32_to_bf16(f - hf);
}

// ---- gfx1250 async global->LDS copy (ASYNCcnt) -----------------------------
#if __has_builtin(__builtin_amdgcn_global_load_async_to_lds_b128)
// Builtin signature (from hipcc diagnostic): param 1 is a pointer to
// __attribute__((vector_size(16))) int in the global address space.
typedef int v4i_vs __attribute__((vector_size(16)));
typedef __attribute__((address_space(1))) v4i_vs GVec;
typedef __attribute__((address_space(3))) v4i_vs LVec;
__device__ __forceinline__ void async_cp32(void* lds, const void* g) {
    GVec* gp = (GVec*)const_cast<void*>(g);
    LVec* lp = (LVec*)lds;
    __builtin_amdgcn_global_load_async_to_lds_b128(gp, lp, 0, 0);
    __builtin_amdgcn_global_load_async_to_lds_b128(gp, lp, 16, 0);
}
__device__ __forceinline__ void async_wait() {
#if __has_builtin(__builtin_amdgcn_s_wait_asynccnt)
    __builtin_amdgcn_s_wait_asynccnt(0);
#else
    asm volatile("s_wait_asynccnt 0" ::: "memory");
#endif
}
#else
// fallback: synchronous copy through VGPRs (still zero VALU converts)
__device__ __forceinline__ void async_cp32(void* lds, const void* g) {
    ((uint4*)lds)[0] = ((const uint4*)g)[0];
    ((uint4*)lds)[1] = ((const uint4*)g)[1];
}
__device__ __forceinline__ void async_wait() {}
#endif

// ---- WMMA fragment loaders (CDNA5 16-bit layouts, CDNA5 ISA) ----
// Tile rows: [row][K..] bf16, K contiguous.  Per 32-K sub-row: 4 uint4 units,
// unit j = K 8j..8j+7.
// A (16x32): lanes 0-15: M=lane, K={0..7,16..23}; lanes 16-31: M=lane-16, K={8..15,24..31}
__device__ __forceinline__ v16bf load_fragA64(const unsigned short* base, int rowOff,
                                              int lane, int ks) {  // row stride 64
    int r = lane & 15, half = lane >> 4;
    const uint4* p = (const uint4*)(base + (rowOff + r) * 64);
    union { uint4 q[2]; v16bf v; } u;
    u.q[0] = p[ks * 4 + half];
    u.q[1] = p[ks * 4 + 2 + half];
    return u.v;
}
__device__ __forceinline__ v16bf load_fragA32(const unsigned short* base, int rowOff,
                                              int lane) {          // row stride 32
    int r = lane & 15, half = lane >> 4;
    const uint4* p = (const uint4*)(base + (rowOff + r) * 32);
    union { uint4 q[2]; v16bf v; } u;
    u.q[0] = p[half];
    u.q[1] = p[2 + half];
    return u.v;
}
// B (32x16), N-major rows: lanes 0-15: N=lane, K={0..7,16..23}; lanes 16-31: N=lane-16, K={8..15,24..31}
__device__ __forceinline__ v16bf load_fragB(const unsigned short* base, int rowOff, int lane) {
    int n = lane & 15, half = lane >> 4;
    const uint4* p = (const uint4*)(base + (rowOff + n) * 32);
    union { uint4 q[2]; v16bf v; } u;
    u.q[0] = p[half];
    u.q[1] = p[2 + half];
    return u.v;
}

#define WMMA_BF16(A, B, C) \
    __builtin_amdgcn_wmma_f32_16x16x32_bf16(false, (A), false, (B), (short)0, (C), false, false)

// ---------------------------------------------------------------------------
// Kernel 1: gating.  One wave per token.
// ---------------------------------------------------------------------------
__global__ __launch_bounds__(256) void moe_gate_kernel(
    const float* __restrict__ x, const float* __restrict__ Wg,
    const float* __restrict__ bg, float* __restrict__ scores,
    int* __restrict__ idxOut)
{
    int lane = threadIdx.x & 31;
    int wid  = threadIdx.x >> 5;
    int row  = blockIdx.x * 8 + wid;
    if (row >= N_TOK) return;

    float p[N_EXP];
#pragma unroll
    for (int e = 0; e < N_EXP; ++e) p[e] = 0.0f;

    const float* xr = x + (size_t)row * D_IN;
    for (int i = lane; i < D_IN; i += 32) {
        float xv = xr[i];
        const float4* w = (const float4*)(Wg + (size_t)i * N_EXP);
        float4 w0 = w[0], w1 = w[1];
        p[0] += xv * w0.x; p[1] += xv * w0.y; p[2] += xv * w0.z; p[3] += xv * w0.w;
        p[4] += xv * w1.x; p[5] += xv * w1.y; p[6] += xv * w1.z; p[7] += xv * w1.w;
    }
#pragma unroll
    for (int off = 16; off >= 1; off >>= 1)
#pragma unroll
        for (int e = 0; e < N_EXP; ++e) p[e] += __shfl_xor(p[e], off, 32);

    if (lane == 0) {
        float g[N_EXP];
        float mx = -3.402823466e38f;
#pragma unroll
        for (int e = 0; e < N_EXP; ++e) { g[e] = p[e] + bg[e]; mx = fmaxf(mx, g[e]); }
        float s = 0.0f;
#pragma unroll
        for (int e = 0; e < N_EXP; ++e) { g[e] = __expf(g[e] - mx); s += g[e]; }
        float inv = 1.0f / s;
#pragma unroll
        for (int e = 0; e < N_EXP; ++e) g[e] *= inv;
        int i1 = 0; float v1 = g[0];
#pragma unroll
        for (int e = 1; e < N_EXP; ++e) if (g[e] > v1) { v1 = g[e]; i1 = e; }
        int i2 = -1; float v2 = -3.402823466e38f;
#pragma unroll
        for (int e = 0; e < N_EXP; ++e) if (e != i1 && g[e] > v2) { v2 = g[e]; i2 = e; }
        scores[row * 2 + 0] = v1;
        scores[row * 2 + 1] = v2;
        if (row == 0) { idxOut[0] = i1; idxOut[1] = i2; }
    }
}

// ---------------------------------------------------------------------------
// Kernel 2: one-time W split-convert into tiled layout:
//   dst[((e*64 + kc)*D_HID + o)*32 + kk] , kc = k/32, kk = k%32
// Rows of 32 K-values are 64B contiguous; a (kc, o-block of 128) B tile is a
// flat 8KB region -> async-copyable straight into LDS.
// ---------------------------------------------------------------------------
__global__ __launch_bounds__(256) void moe_wconv_kernel(
    const float* __restrict__ We, const int* __restrict__ idx,
    unsigned short* __restrict__ wHi, unsigned short* __restrict__ wLo)
{
    int gid = blockIdx.x * 256 + threadIdx.x;   // 0 .. 2*64*2048-1
    int o   = gid & (D_HID - 1);
    int kc  = (gid >> 11) & 63;
    int e   = gid >> 17;
    int ex  = idx[e];

    const float* src = We + ((size_t)ex * D_IN + (size_t)kc * 32) * D_HID + o;
    unsigned hw[16], lw[16];
#pragma unroll
    for (int j = 0; j < 16; ++j) {
        unsigned short h0, l0, h1, l1;
        cvt_split(src[(size_t)(2 * j)     * D_HID], h0, l0);
        cvt_split(src[(size_t)(2 * j + 1) * D_HID], h1, l1);
        hw[j] = (unsigned)h0 | ((unsigned)h1 << 16);
        lw[j] = (unsigned)l0 | ((unsigned)l1 << 16);
    }
    size_t base = ((size_t)(e * 64 + kc) * D_HID + o) * 32;
    uint4* dh = (uint4*)(wHi + base);
    uint4* dl = (uint4*)(wLo + base);
#pragma unroll
    for (int q = 0; q < 4; ++q) {
        dh[q] = make_uint4(hw[4*q], hw[4*q+1], hw[4*q+2], hw[4*q+3]);
        dl[q] = make_uint4(lw[4*q], lw[4*q+1], lw[4*q+2], lw[4*q+3]);
    }
}

// ---------------------------------------------------------------------------
// Shared epilogue: out[n,o] = s0[n]*(acc0+b0[o]) + s1[n]*(acc1+b1[o])
// C layout: lanes 0-15: (M=r, N=lane); lanes 16-31: (M=8+r, N=lane-16)
// ---------------------------------------------------------------------------
__device__ __forceinline__ void moe_epilogue(
    const v8f acc0[2][4], const v8f acc1[2][4],
    const float* be, const float* scores, float* out,
    int e0, int e1, int oBase, int mBase, int waveM, int waveO, int lane)
{
    const float* b0 = be + (size_t)e0 * D_HID;
    const float* b1 = be + (size_t)e1 * D_HID;
    const int Nc    = lane & 15;
    const int Mhalf = (lane >> 4) * 8;
#pragma unroll
    for (int mt = 0; mt < 2; ++mt)
#pragma unroll
        for (int nt = 0; nt < 4; ++nt) {
            int col = oBase + waveO + nt * 16 + Nc;
            float bb0 = b0[col], bb1 = b1[col];
            int rowBase = mBase + waveM + mt * 16 + Mhalf;
#pragma unroll
            for (int r = 0; r < 8; ++r) {
                int row = rowBase + r;
                float s0 = scores[row * 2 + 0];
                float s1 = scores[row * 2 + 1];
                out[(size_t)row * D_HID + col] =
                    s0 * (acc0[mt][nt][r] + bb0) + s1 * (acc1[mt][nt][r] + bb1);
            }
        }
}

// ---------------------------------------------------------------------------
// Kernel 3 (primary): fused 2-expert GEMM, pre-converted W, async B staging.
// Block = 256 threads (8 waves) -> 128(M) x 128(O) tile; K-chunk = 64.
// LDS: A 32KB + B 64KB = 96KB (of 320KB/WGP).
// ---------------------------------------------------------------------------
__global__ __launch_bounds__(256) void moe_wmma_pre_kernel(
    const float* __restrict__ x,
    const unsigned short* __restrict__ wHi, const unsigned short* __restrict__ wLo,
    const float* __restrict__ be, const float* __restrict__ scores,
    const int* __restrict__ idx, float* __restrict__ out)
{
    __shared__ __align__(16) unsigned short sAhi[128 * 64];
    __shared__ __align__(16) unsigned short sAlo[128 * 64];
    __shared__ __align__(16) unsigned short sBhi[2][2][128 * 32];  // [expert][ks]
    __shared__ __align__(16) unsigned short sBlo[2][2][128 * 32];

    const int oBase = blockIdx.x * 128;
    const int mBase = blockIdx.y * 128;

    const int t    = threadIdx.x;
    const int lane = t & 31;
    const int wid  = t >> 5;
    const int waveM = (wid & 3) * 32;
    const int waveO = (wid >> 2) * 64;

    const int e0 = idx[0];
    const int e1 = idx[1];

    v8f acc0[2][4], acc1[2][4];
#pragma unroll
    for (int mt = 0; mt < 2; ++mt)
#pragma unroll
        for (int nt = 0; nt < 4; ++nt) { acc0[mt][nt] = (v8f){}; acc1[mt][nt] = (v8f){}; }

    const int am = t >> 1;              // A row 0..127
    const int ak = (t & 1) * 32;        // A k-offset 0/32 within chunk

    for (int k0 = 0; k0 < D_IN; k0 += 64) {
        // ---- B: async copy 8 pre-tiled 8KB sub-tiles into LDS ----
        const int kc = k0 >> 5;
#pragma unroll
        for (int e = 0; e < 2; ++e)
#pragma unroll
            for (int ks = 0; ks < 2; ++ks) {
                size_t gbase = ((size_t)(e * 64 + kc + ks) * D_HID + oBase) * 32;
                async_cp32((char*)sBhi[e][ks] + t * 32,
                           (const char*)(wHi + gbase) + t * 32);
                async_cp32((char*)sBlo[e][ks] + t * 32,
                           (const char*)(wLo + gbase) + t * 32);
            }
        // ---- A: load fp32, split to bf16 hi/lo, store to LDS ----
        {
            const float* src = x + (size_t)(mBase + am) * D_IN + k0 + ak;
            if (k0 + 64 < D_IN) __builtin_prefetch(src + 64, 0, 3);  // global_prefetch_b8
            float v[32];
#pragma unroll
            for (int q = 0; q < 8; ++q) {
                float4 f = ((const float4*)src)[q];
                v[4*q] = f.x; v[4*q+1] = f.y; v[4*q+2] = f.z; v[4*q+3] = f.w;
            }
            unsigned hw[16], lw[16];
#pragma unroll
            for (int j = 0; j < 16; ++j) {
                unsigned short h0, l0, h1, l1;
                cvt_split(v[2*j],   h0, l0);
                cvt_split(v[2*j+1], h1, l1);
                hw[j] = (unsigned)h0 | ((unsigned)h1 << 16);
                lw[j] = (unsigned)l0 | ((unsigned)l1 << 16);
            }
            uint4* dh = (uint4*)(sAhi + am * 64 + ak);
            uint4* dl = (uint4*)(sAlo + am * 64 + ak);
#pragma unroll
            for (int q = 0; q < 4; ++q) {
                dh[q] = make_uint4(hw[4*q], hw[4*q+1], hw[4*q+2], hw[4*q+3]);
                dl[q] = make_uint4(lw[4*q], lw[4*q+1], lw[4*q+2], lw[4*q+3]);
            }
        }
        async_wait();          // s_wait_asynccnt 0
        __syncthreads();

        // ---- compute: 96 WMMAs per wave per chunk ----
#pragma unroll
        for (int ks = 0; ks < 2; ++ks) {
            v16bf aH[2], aL[2];
#pragma unroll
            for (int mt = 0; mt < 2; ++mt) {
                aH[mt] = load_fragA64(sAhi, waveM + mt * 16, lane, ks);
                aL[mt] = load_fragA64(sAlo, waveM + mt * 16, lane, ks);
            }
#pragma unroll
            for (int nt = 0; nt < 4; ++nt) {
                int ro = waveO + nt * 16;
                v16bf b0h = load_fragB(sBhi[0][ks], ro, lane);
                v16bf b0l = load_fragB(sBlo[0][ks], ro, lane);
                v16bf b1h = load_fragB(sBhi[1][ks], ro, lane);
                v16bf b1l = load_fragB(sBlo[1][ks], ro, lane);
#pragma unroll
                for (int mt = 0; mt < 2; ++mt) {
                    acc0[mt][nt] = WMMA_BF16(aH[mt], b0h, acc0[mt][nt]);
                    acc0[mt][nt] = WMMA_BF16(aH[mt], b0l, acc0[mt][nt]);
                    acc0[mt][nt] = WMMA_BF16(aL[mt], b0h, acc0[mt][nt]);
                    acc1[mt][nt] = WMMA_BF16(aH[mt], b1h, acc1[mt][nt]);
                    acc1[mt][nt] = WMMA_BF16(aH[mt], b1l, acc1[mt][nt]);
                    acc1[mt][nt] = WMMA_BF16(aL[mt], b1h, acc1[mt][nt]);
                }
            }
        }
        __syncthreads();
    }

    moe_epilogue(acc0, acc1, be, scores, out, e0, e1, oBase, mBase, waveM, waveO, lane);
}

// ---------------------------------------------------------------------------
// Kernel 3b (fallback if ws too small): round-1 style, converts W in-loop.
// ---------------------------------------------------------------------------
__global__ __launch_bounds__(256) void moe_wmma_fused_kernel(
    const float* __restrict__ x, const float* __restrict__ We,
    const float* __restrict__ be, const float* __restrict__ scores,
    const int* __restrict__ idx, float* __restrict__ out)
{
    __shared__ __align__(16) unsigned short sAhi[128 * 32];
    __shared__ __align__(16) unsigned short sAlo[128 * 32];
    __shared__ __align__(16) unsigned short sBhi[2][128 * 32];
    __shared__ __align__(16) unsigned short sBlo[2][128 * 32];

    const int oBase = blockIdx.x * 128;
    const int mBase = blockIdx.y * 128;
    const int t    = threadIdx.x;
    const int lane = t & 31;
    const int wid  = t >> 5;
    const int waveM = (wid & 3) * 32;
    const int waveO = (wid >> 2) * 64;

    const int e0 = idx[0];
    const int e1 = idx[1];
    const float* W0 = We + (size_t)e0 * D_IN * D_HID;
    const float* W1 = We + (size_t)e1 * D_IN * D_HID;

    v8f acc0[2][4], acc1[2][4];
#pragma unroll
    for (int mt = 0; mt < 2; ++mt)
#pragma unroll
        for (int nt = 0; nt < 4; ++nt) { acc0[mt][nt] = (v8f){}; acc1[mt][nt] = (v8f){}; }

    const int am  = t >> 1;
    const int akh = (t & 1) * 16;
    const int bo  = t & 127;
    const int bkh = (t >> 7) * 16;

    for (int k0 = 0; k0 < D_IN; k0 += 32) {
        {
            const float* src = x + (size_t)(mBase + am) * D_IN + k0 + akh;
            float v[16];
#pragma unroll
            for (int q = 0; q < 4; ++q) {
                float4 f = ((const float4*)src)[q];
                v[4*q] = f.x; v[4*q+1] = f.y; v[4*q+2] = f.z; v[4*q+3] = f.w;
            }
            unsigned hw[8], lw[8];
#pragma unroll
            for (int j = 0; j < 8; ++j) {
                unsigned short h0, l0, h1, l1;
                cvt_split(v[2*j], h0, l0);
                cvt_split(v[2*j+1], h1, l1);
                hw[j] = (unsigned)h0 | ((unsigned)h1 << 16);
                lw[j] = (unsigned)l0 | ((unsigned)l1 << 16);
            }
            uint4* dh = (uint4*)(sAhi + am * 32 + akh);
            uint4* dl = (uint4*)(sAlo + am * 32 + akh);
            dh[0] = make_uint4(hw[0], hw[1], hw[2], hw[3]);
            dh[1] = make_uint4(hw[4], hw[5], hw[6], hw[7]);
            dl[0] = make_uint4(lw[0], lw[1], lw[2], lw[3]);
            dl[1] = make_uint4(lw[4], lw[5], lw[6], lw[7]);
        }
#pragma unroll
        for (int e = 0; e < 2; ++e) {
            const float* W = (e == 0) ? W0 : W1;
            const float* src = W + (size_t)(k0 + bkh) * D_HID + oBase + bo;
            float v[16];
#pragma unroll
            for (int kk = 0; kk < 16; ++kk) v[kk] = src[(size_t)kk * D_HID];
            unsigned hw[8], lw[8];
#pragma unroll
            for (int j = 0; j < 8; ++j) {
                unsigned short h0, l0, h1, l1;
                cvt_split(v[2*j], h0, l0);
                cvt_split(v[2*j+1], h1, l1);
                hw[j] = (unsigned)h0 | ((unsigned)h1 << 16);
                lw[j] = (unsigned)l0 | ((unsigned)l1 << 16);
            }
            uint4* dh = (uint4*)(sBhi[e] + bo * 32 + bkh);
            uint4* dl = (uint4*)(sBlo[e] + bo * 32 + bkh);
            dh[0] = make_uint4(hw[0], hw[1], hw[2], hw[3]);
            dh[1] = make_uint4(hw[4], hw[5], hw[6], hw[7]);
            dl[0] = make_uint4(lw[0], lw[1], lw[2], lw[3]);
            dl[1] = make_uint4(lw[4], lw[5], lw[6], lw[7]);
        }
        __syncthreads();

        v16bf aH[2], aL[2];
#pragma unroll
        for (int mt = 0; mt < 2; ++mt) {
            aH[mt] = load_fragA32(sAhi, waveM + mt * 16, lane);
            aL[mt] = load_fragA32(sAlo, waveM + mt * 16, lane);
        }
#pragma unroll
        for (int nt = 0; nt < 4; ++nt) {
            int ro = waveO + nt * 16;
            v16bf b0h = load_fragB(sBhi[0], ro, lane);
            v16bf b0l = load_fragB(sBlo[0], ro, lane);
            v16bf b1h = load_fragB(sBhi[1], ro, lane);
            v16bf b1l = load_fragB(sBlo[1], ro, lane);
#pragma unroll
            for (int mt = 0; mt < 2; ++mt) {
                acc0[mt][nt] = WMMA_BF16(aH[mt], b0h, acc0[mt][nt]);
                acc0[mt][nt] = WMMA_BF16(aH[mt], b0l, acc0[mt][nt]);
                acc0[mt][nt] = WMMA_BF16(aL[mt], b0h, acc0[mt][nt]);
                acc1[mt][nt] = WMMA_BF16(aH[mt], b1h, acc1[mt][nt]);
                acc1[mt][nt] = WMMA_BF16(aH[mt], b1l, acc1[mt][nt]);
                acc1[mt][nt] = WMMA_BF16(aL[mt], b1h, acc1[mt][nt]);
            }
        }
        __syncthreads();
    }

    moe_epilogue(acc0, acc1, be, scores, out, e0, e1, oBase, mBase, waveM, waveO, lane);
}

// ---------------------------------------------------------------------------
extern "C" void kernel_launch(void* const* d_in, const int* in_sizes, int n_in,
                              void* d_out, int out_size, void* d_ws, size_t ws_size,
                              hipStream_t stream) {
    const float* x  = (const float*)d_in[0];   // [N, D_IN]
    const float* We = (const float*)d_in[1];   // [E, D_IN, D_HID]
    const float* be = (const float*)d_in[2];   // [E, D_HID]
    const float* Wg = (const float*)d_in[3];   // [D_IN, E]
    const float* bg = (const float*)d_in[4];   // [E]
    float* out = (float*)d_out;                // [N, D_HID]

    int*   idxp   = (int*)d_ws;
    float* scores = (float*)((char*)d_ws + WS_SCORE_OFF);

    moe_gate_kernel<<<N_TOK / 8, 256, 0, stream>>>(x, Wg, bg, scores, idxp);

    if (ws_size >= WS_NEED) {
        unsigned short* wHi = (unsigned short*)((char*)d_ws + WS_WHI_OFF);
        unsigned short* wLo = (unsigned short*)((char*)d_ws + WS_WLO_OFF);
        moe_wconv_kernel<<<(2 * 64 * D_HID) / 256, 256, 0, stream>>>(We, idxp, wHi, wLo);
        moe_wmma_pre_kernel<<<dim3(D_HID / 128, N_TOK / 128), 256, 0, stream>>>(
            x, wHi, wLo, be, scores, idxp, out);
    } else {
        moe_wmma_fused_kernel<<<dim3(D_HID / 128, N_TOK / 128), 256, 0, stream>>>(
            x, We, be, scores, idxp, out);
    }
}
